// GraphSAGE_post_64630667870460
// MI455X (gfx1250) — matrix operand order repaired and checked by reading, not compiled, not run on hardware
//
#include <hip/hip_runtime.h>
#include <hip/hip_bf16.h>

typedef __attribute__((ext_vector_type(16))) __bf16 v16bf;
typedef __attribute__((ext_vector_type(8)))  __bf16 v8bf;
typedef __attribute__((ext_vector_type(8)))  float  v8f;

// ---------------------------------------------------------------------------
// f32 -> bf16 elementwise convert
// ---------------------------------------------------------------------------
__global__ void k_f32_to_bf16(const float* __restrict__ s, __bf16* __restrict__ d, int n) {
  int i = blockIdx.x * blockDim.x + threadIdx.x;
  if (i < n) d[i] = (__bf16)s[i];
}

// ---------------------------------------------------------------------------
// W [din][dout] f32 row-major -> WT [dout][din] bf16  (B operand, contiguous-K)
// ---------------------------------------------------------------------------
__global__ void k_transpose_bf16(const float* __restrict__ W, __bf16* __restrict__ WT,
                                 int din, int dout) {
  int i = blockIdx.x * blockDim.x + threadIdx.x;
  if (i >= din * dout) return;
  int n = i / din;
  int k = i - n * din;
  WT[i] = (__bf16)W[(size_t)k * dout + n];
}

// ---------------------------------------------------------------------------
// Edge scatter-max: pooled[dst] = max(pooled[dst], m[src]) per feature.
// m >= 0 (relu), pooled initialized to 0 -> u32 bit-pattern atomic max is exact
// and empty neighborhoods stay 0 (matches reference's isneginf->0).
// One thread = one edge x 4 features (float4). Consecutive lanes cover one
// edge's contiguous feature row -> coalesced gather + coalesced atomics.
// ---------------------------------------------------------------------------
__global__ void k_scatter_max(const float4* __restrict__ m, const int* __restrict__ srcIdx,
                              const int* __restrict__ dstIdx, unsigned int* __restrict__ pooled,
                              int nEdges, int cpeShift) {
  unsigned tid = blockIdx.x * blockDim.x + threadIdx.x;
  unsigned e = tid >> cpeShift;
  if (e >= (unsigned)nEdges) return;
  unsigned cpe = 1u << cpeShift;
  unsigned c = tid & (cpe - 1u);
  int s = srcIdx[e];
  int d = dstIdx[e];
  float4 v = m[(size_t)s * cpe + c];
  unsigned int* p = pooled + (((size_t)d * cpe + c) << 2);
  atomicMax(p + 0, __float_as_uint(v.x));
  atomicMax(p + 1, __float_as_uint(v.y));
  atomicMax(p + 2, __float_as_uint(v.z));
  atomicMax(p + 3, __float_as_uint(v.w));
}

// ---------------------------------------------------------------------------
// Fused WMMA GEMM: out = act( A1 @ B1 + bias [+ A2 @ B2] )
//  A*  : [rows][K] bf16 row-major
//  B*T : [Nout][K] bf16 (pre-transposed weights, so B-fragment load is
//        two 16B reads per lane)
//  out : [rows][Nout] f32
// One wave32 per 16x16 output tile; K stepped by 32 via
// v_wmma_f32_16x16x32_bf16 with f32 accumulation.
// ---------------------------------------------------------------------------
__global__ void __launch_bounds__(32)
k_wmma_gemm(const __bf16* __restrict__ A1, const __bf16* __restrict__ B1T,
            const __bf16* __restrict__ A2, const __bf16* __restrict__ B2T,
            const float* __restrict__ bias, float* __restrict__ out,
            int K, int Nout, int doRelu) {
  const int lane  = threadIdx.x & 31;
  const int r     = lane & 15;
  const int half  = lane >> 4;       // 0: lanes 0-15, 1: lanes 16-31
  const int mtile = blockIdx.x << 4;
  const int ntile = blockIdx.y << 4;

  // ISA 16-bit A layout: lane half selects K offset {0,8} inside each 16-block
  const size_t aRow = (size_t)(mtile + r) * K;
  const int    kA   = half * 8;
  // B layout matches A: lane half selects K offset {0,8} inside each 16-block
  const size_t bRow = (size_t)(ntile + r) * K;
  const int    kB   = half * 8;

  v8f acc = {};
  for (int k0 = 0; k0 < K; k0 += 32) {
    v8bf a0 = *(const v8bf*)(A1 + aRow + k0 + kA);
    v8bf a1 = *(const v8bf*)(A1 + aRow + k0 + 16 + kA);
    v16bf a = __builtin_shufflevector(a0, a1, 0, 1, 2, 3, 4, 5, 6, 7,
                                              8, 9, 10, 11, 12, 13, 14, 15);
    v8bf b0 = *(const v8bf*)(B1T + bRow + k0 + kB);
    v8bf b1 = *(const v8bf*)(B1T + bRow + k0 + 16 + kB);
    v16bf b = __builtin_shufflevector(b0, b1, 0, 1, 2, 3, 4, 5, 6, 7,
                                              8, 9, 10, 11, 12, 13, 14, 15);
    acc = __builtin_amdgcn_wmma_f32_16x16x32_bf16(false, a, false, b,
                                                  (short)0, acc, false, false);
  }
  if (A2 != nullptr) {
    for (int k0 = 0; k0 < K; k0 += 32) {
      v8bf a0 = *(const v8bf*)(A2 + aRow + k0 + kA);
      v8bf a1 = *(const v8bf*)(A2 + aRow + k0 + 16 + kA);
      v16bf a = __builtin_shufflevector(a0, a1, 0, 1, 2, 3, 4, 5, 6, 7,
                                                8, 9, 10, 11, 12, 13, 14, 15);
      v8bf b0 = *(const v8bf*)(B2T + bRow + k0 + kB);
      v8bf b1 = *(const v8bf*)(B2T + bRow + k0 + 16 + kB);
      v16bf b = __builtin_shufflevector(b0, b1, 0, 1, 2, 3, 4, 5, 6, 7,
                                                8, 9, 10, 11, 12, 13, 14, 15);
      acc = __builtin_amdgcn_wmma_f32_16x16x32_bf16(false, a, false, b,
                                                    (short)0, acc, false, false);
    }
  }

  // C/D layout: VGPR v, lanes 0-15 -> row v, lanes 16-31 -> row v+8; col = lane&15
  const int n    = ntile + r;
  const int mrow = mtile + half * 8;
  const float bv = bias ? bias[n] : 0.0f;
#pragma unroll
  for (int v = 0; v < 8; ++v) {
    float x = acc[v] + bv;
    if (doRelu) x = fmaxf(x, 0.0f);
    out[(size_t)(mrow + v) * Nout + n] = x;
  }
}

// ---------------------------------------------------------------------------
// Host-side orchestration
// ---------------------------------------------------------------------------
extern "C" void kernel_launch(void* const* d_in, const int* in_sizes, int n_in,
                              void* d_out, int out_size, void* d_ws, size_t ws_size,
                              hipStream_t stream) {
  const int IN_FEATS = 128;
  const int N = in_sizes[0] / IN_FEATS;   // 50000 (multiple of 16)
  const int E = in_sizes[1];              // 800000

  const float* feats = (const float*)d_in[0];
  const int* srcIdx  = (const int*)d_in[1];
  const int* dstIdx  = (const int*)d_in[2];

  // Layer dims: (din, dout)
  const int di_[3] = {128, 128, 64};
  const int do_[3] = {128, 64, 32};

  // Param flattening order: dict-insertion [Wp,bp,Ws,bs,Wn] vs
  // pytree-alphabetical [Wn,Wp,Ws,bp,bs]; detect via layer-0 second entry.
  const bool insertion = (in_sizes[4] == 128);
  const float *Wp[3], *bp[3], *Ws[3], *bs[3], *Wn[3];
  for (int l = 0; l < 3; ++l) {
    int b = 3 + 5 * l;
    if (insertion) {
      Wp[l] = (const float*)d_in[b + 0];
      bp[l] = (const float*)d_in[b + 1];
      Ws[l] = (const float*)d_in[b + 2];
      bs[l] = (const float*)d_in[b + 3];
      Wn[l] = (const float*)d_in[b + 4];
    } else {
      Wn[l] = (const float*)d_in[b + 0];
      Wp[l] = (const float*)d_in[b + 1];
      Ws[l] = (const float*)d_in[b + 2];
      bp[l] = (const float*)d_in[b + 3];
      bs[l] = (const float*)d_in[b + 4];
    }
  }

  // Workspace carve-out (256B aligned)
  char* ws = (char*)d_ws;
  size_t off = 0;
  auto take = [&](size_t bytes) -> char* {
    char* p = ws + off;
    off += (bytes + 255) & ~(size_t)255;
    return p;
  };
  __bf16* hbf      = (__bf16*)take((size_t)N * 128 * sizeof(__bf16));
  float*  mbuf     = (float*) take((size_t)N * 128 * sizeof(float));
  float*  pooled   = (float*) take((size_t)N * 128 * sizeof(float));
  __bf16* pooledbf = (__bf16*)take((size_t)N * 128 * sizeof(__bf16));
  float*  h1       = (float*) take((size_t)N * 128 * sizeof(float));
  __bf16* wt       = (__bf16*)take((size_t)100000 * sizeof(__bf16));

  // Slice + transpose-convert all weights to bf16 [dout][din]
  __bf16 *WpT[3], *WsT[3], *WnT[3];
  size_t wc = 0;
  for (int l = 0; l < 3; ++l) {
    WpT[l] = wt + wc; wc += (size_t)di_[l] * di_[l];
    WsT[l] = wt + wc; wc += (size_t)di_[l] * do_[l];
    WnT[l] = wt + wc; wc += (size_t)di_[l] * do_[l];
  }
  for (int l = 0; l < 3; ++l) {
    int np = di_[l] * di_[l];
    int ns = di_[l] * do_[l];
    k_transpose_bf16<<<(np + 255) / 256, 256, 0, stream>>>(Wp[l], WpT[l], di_[l], di_[l]);
    k_transpose_bf16<<<(ns + 255) / 256, 256, 0, stream>>>(Ws[l], WsT[l], di_[l], do_[l]);
    k_transpose_bf16<<<(ns + 255) / 256, 256, 0, stream>>>(Wn[l], WnT[l], di_[l], do_[l]);
  }

  float* aspectOut = (float*)d_out;            // [N, 64]
  float* finalOut  = (float*)d_out + (size_t)N * 64;  // [N, 32]

  const float* h = feats;
  for (int l = 0; l < 3; ++l) {
    const int di = di_[l], dout = do_[l];
    const int nh = N * di;

    // h (f32) -> bf16 operand
    k_f32_to_bf16<<<(nh + 255) / 256, 256, 0, stream>>>(h, hbf, nh);

    // m = relu(h @ Wp + bp)  [N, di]
    {
      dim3 g(N / 16, di / 16);
      k_wmma_gemm<<<g, 32, 0, stream>>>(hbf, WpT[l], nullptr, nullptr,
                                        bp[l], mbuf, di, di, 1);
    }

    // pooled = segment_max(m[src], dst), empty -> 0
    hipMemsetAsync(pooled, 0, (size_t)nh * sizeof(float), stream);
    {
      const int cpeShift = (di == 128) ? 5 : 4;   // float4 chunks per edge
      const long long total = (long long)E << cpeShift;
      const int blocks = (int)((total + 255) / 256);
      k_scatter_max<<<blocks, 256, 0, stream>>>((const float4*)mbuf, srcIdx, dstIdx,
                                                (unsigned int*)pooled, E, cpeShift);
    }
    k_f32_to_bf16<<<(nh + 255) / 256, 256, 0, stream>>>(pooled, pooledbf, nh);

    // h' = act( h @ Ws + bs + pooled @ Wn )
    float* outp = (l == 0) ? h1 : (l == 1) ? aspectOut : finalOut;
    {
      dim3 g(N / 16, dout / 16);
      k_wmma_gemm<<<g, 32, 0, stream>>>(hbf, WsT[l], pooledbf, WnT[l],
                                        bs[l], outp, di, dout, (l == 0) ? 1 : 0);
    }
    h = outp;
  }
}
